// ChannelWiseRowAttention_26268019982927
// MI455X (gfx1250) — matrix operand order repaired and checked by reading, not compiled, not run on hardware
//
#include <hip/hip_runtime.h>
#include <hip/hip_bf16.h>

typedef __attribute__((ext_vector_type(16))) _Float16 v16h;
typedef __attribute__((ext_vector_type(8)))  float    v8f;

#define N_  2
#define C_  128
#define H_  512
#define W_  256
#define BQ  128     // query rows per workgroup (8 waves x 16)
#define BK  64      // key/value rows per inner block
#define NWAVE 8
#define LDK 136     // padded LDS row stride (halfs) for K tile   (rows 272B -> 16B aligned)
#define LDV 72      // padded LDS row stride (halfs) for V^T tile (rows 144B -> 16B aligned)
#define LDP 72      // padded LDS row stride (halfs) for P staging

// ---- fragment loaders (CDNA5 WMMA 16-bit layouts, ISA 7.12.2) ----

// A-matrix 16x32 f16: lane holds row m=lane%16; lanes>=16 hold K-group +8;
// VGPR v packs K = (v/4)*16 + kgrp + (v%4)*2 , +1  -> two contiguous 16B chunks
__device__ __forceinline__ v16h load_a_frag(const _Float16* S, int ld, int k0, int lane) {
    const int m  = lane & 15;
    const int kg = (lane >> 4) * 8;
    const _Float16* p = S + m * ld + k0;
    v16h a;
#pragma unroll
    for (int vv = 0; vv < 8; ++vv) {
        const int kk = (vv >> 2) * 16 + kg + (vv & 3) * 2;
        a[2 * vv]     = p[kk];
        a[2 * vv + 1] = p[kk + 1];
    }
    return a;
}

// B-matrix 32x16 f16 where the logical B is S^T: B[k][n] = S[n0+n][k0+k].
// Lane holds col n=lane%16; lanes>=16 hold K-group +8; elems 0-7 K=kgrp+s, 8-15 K=16+kgrp+(s-8).
__device__ __forceinline__ v16h load_bT_frag(const _Float16* S, int ld, int n0, int k0, int lane) {
    const int n  = lane & 15;
    const int kg = (lane >> 4) * 8;
    const _Float16* p = S + (n0 + n) * ld + k0 + kg;
    v16h b;
#pragma unroll
    for (int vv = 0; vv < 8; ++vv) { b[vv] = p[vv]; b[8 + vv] = p[16 + vv]; }
    return b;
}

__global__ __launch_bounds__(256) void
ChannelWiseRowAttention_fa_kernel(const float* __restrict__ q,
                                  const float* __restrict__ k,
                                  const float* __restrict__ v,
                                  float* __restrict__ out) {
    const int tid  = threadIdx.x;
    const int lane = tid & 31;
    const int wave = tid >> 5;
    const int w    = blockIdx.x;           // fastest dim -> w-neighbors share L2 lines
    const int q0   = blockIdx.y * BQ;
    const int n    = blockIdx.z;

    const size_t base = (size_t)n * C_ * H_ * W_ + (size_t)w;
    const size_t cs   = (size_t)H_ * W_;   // stride along c; element (h,c) = base + c*cs + h*W_

    __shared__ _Float16 Ks[BK * LDK];      // K block, row-major [key][c]
    __shared__ _Float16 Vt[C_ * LDV];      // V block, transposed [c][key]
    __shared__ _Float16 Ps[NWAVE * 16 * LDP];

    // ---- Q A-fragments for this wave's 16 rows, straight from global ----
    const int am  = lane & 15;
    const int akg = (lane >> 4) * 8;
    const float* qrow = q + base + (size_t)(q0 + wave * 16 + am) * W_;
    v16h aq[4];
#pragma unroll
    for (int kt = 0; kt < 4; ++kt) {
        v16h a;
#pragma unroll
        for (int vv = 0; vv < 8; ++vv) {
            const int ck = kt * 32 + (vv >> 2) * 16 + akg + (vv & 3) * 2;
            a[2 * vv]     = (_Float16)qrow[(size_t)ck * cs];
            a[2 * vv + 1] = (_Float16)qrow[(size_t)(ck + 1) * cs];
        }
        aq[kt] = a;
    }

    // ---- per-wave running state (base-2 softmax domain) ----
    const v8f zero8 = {0.f, 0.f, 0.f, 0.f, 0.f, 0.f, 0.f, 0.f};
    v8f acc[8];
#pragma unroll
    for (int t = 0; t < 8; ++t) acc[t] = zero8;
    float mrow[8], lrow[8];
#pragma unroll
    for (int r = 0; r < 8; ++r) { mrow[r] = -1.0e30f; lrow[r] = 0.f; }

    // fold 1/sqrt(C) and log2(e) together: exp2 of base-2 scores == softmax weights
    const float scale2 = 0.08838834764831845f * 1.4426950408889634f;
    const int colL = lane & 15;
    const int rOff = (lane >> 4) * 8;
    _Float16* Pw = Ps + wave * 16 * LDP;

    for (int kb = 0; kb < H_ / BK; ++kb) {
        const int kr0 = kb * BK;
        __syncthreads();   // previous iteration done reading Ks/Vt
        // cooperative load of K/V block, f32 -> f16; lanes walk h (1KB-spaced lines)
        for (int i = tid; i < BK * C_; i += 256) {
            const int r = i & (BK - 1);
            const int c = i >> 6;
            const size_t off = base + (size_t)c * cs + (size_t)(kr0 + r) * W_;
            Ks[r * LDK + c] = (_Float16)k[off];
            Vt[c * LDV + r] = (_Float16)v[off];
        }
        if (kb + 1 < H_ / BK) {   // touch next block's K lines early
            const int r = tid & (BK - 1);
            const int c = tid >> 6;
            __builtin_prefetch(&k[base + (size_t)c * cs + (size_t)(kr0 + BK + r) * W_], 0, 1);
        }
        __syncthreads();

        // ---- S2 = (Q K^T) * scale2 : 16 x 64 per wave (base-2 domain) ----
        v8f s[4];
#pragma unroll
        for (int jt = 0; jt < 4; ++jt) {
            v8f c0 = zero8;
#pragma unroll
            for (int kt = 0; kt < 4; ++kt) {
                const v16h b = load_bT_frag(Ks, LDK, jt * 16, kt * 32, lane);
                c0 = __builtin_amdgcn_wmma_f32_16x16x32_f16(
                        false, aq[kt], false, b, (short)0, c0, false, false);
            }
            s[jt] = c0 * scale2;
        }

        // ---- online softmax (rows live in 16-lane halves of C layout) ----
#pragma unroll
        for (int r = 0; r < 8; ++r) {
            float mx = fmaxf(fmaxf(s[0][r], s[1][r]), fmaxf(s[2][r], s[3][r]));
#pragma unroll
            for (int msk = 8; msk >= 1; msk >>= 1)
                mx = fmaxf(mx, __shfl_xor(mx, msk, 32));
            const float mnew = fmaxf(mrow[r], mx);
            const float corr = exp2f(mrow[r] - mnew);
            mrow[r] = mnew;
            lrow[r] *= corr;
#pragma unroll
            for (int t = 0; t < 8; ++t) acc[t][r] *= corr;
        }
#pragma unroll
        for (int jt = 0; jt < 4; ++jt)
#pragma unroll
            for (int r = 0; r < 8; ++r)
                s[jt][r] = exp2f(s[jt][r] - mrow[r]);
#pragma unroll
        for (int r = 0; r < 8; ++r) {
            float sum = (s[0][r] + s[1][r]) + (s[2][r] + s[3][r]);
#pragma unroll
            for (int msk = 8; msk >= 1; msk >>= 1)
                sum += __shfl_xor(sum, msk, 32);
            lrow[r] += sum;
        }

        // ---- stage P (C layout -> f16 A layout) through padded LDS ----
#pragma unroll
        for (int jt = 0; jt < 4; ++jt)
#pragma unroll
            for (int r = 0; r < 8; ++r)
                Pw[(r + rOff) * LDP + jt * 16 + colL] = (_Float16)s[jt][r];
        __syncthreads();

        // ---- O += P @ V  (V^T tile -> contiguous B-fragment loads) ----
        v16h pf[2];
#pragma unroll
        for (int kt = 0; kt < 2; ++kt) pf[kt] = load_a_frag(Pw, LDP, kt * 32, lane);
#pragma unroll
        for (int t = 0; t < 8; ++t) {
#pragma unroll
            for (int kt = 0; kt < 2; ++kt) {
                // B[k][n] = V[key=k][c=n] = Vt[c=n][key=k]
                const v16h bv = load_bT_frag(Vt, LDV, t * 16, kt * 32, lane);
                acc[t] = __builtin_amdgcn_wmma_f32_16x16x32_f16(
                            false, pf[kt], false, bv, (short)0, acc[t], false, false);
            }
        }
    }

    // ---- normalize and scatter back to NCHW ----
#pragma unroll
    for (int r = 0; r < 8; ++r) {
        const float inv = 1.0f / lrow[r];
#pragma unroll
        for (int t = 0; t < 8; ++t) acc[t][r] *= inv;
    }
    const int hBase = q0 + wave * 16 + rOff;
#pragma unroll
    for (int t = 0; t < 8; ++t) {
        const int c = t * 16 + colL;
#pragma unroll
        for (int r = 0; r < 8; ++r)
            out[base + (size_t)c * cs + (size_t)(hBase + r) * W_] = acc[t][r];
    }
}

extern "C" void kernel_launch(void* const* d_in, const int* in_sizes, int n_in,
                              void* d_out, int out_size, void* d_ws, size_t ws_size,
                              hipStream_t stream) {
    const float* q = (const float*)d_in[0];
    const float* k = (const float*)d_in[1];
    const float* v = (const float*)d_in[2];
    float* out = (float*)d_out;
    dim3 grid(W_, H_ / BQ, N_);   // w fastest -> L2 cacheline sharing across WGs
    ChannelWiseRowAttention_fa_kernel<<<grid, 256, 0, stream>>>(q, k, v, out);
}
